// SwinTransformerBlock_48352741818781
// MI455X (gfx1250) — compile-verified, matrix operand order checked by reading, not run on hardware
//
#include <hip/hip_runtime.h>
#include <hip/hip_bf16.h>

// ---------------------------------------------------------------------------
// Types for CDNA5 WMMA (wave32, v_wmma_f32_16x16x32_bf16)
// ---------------------------------------------------------------------------
typedef __bf16 bf16_t;
typedef __attribute__((ext_vector_type(16))) __bf16 v16bf;
typedef __attribute__((ext_vector_type(8)))  __bf16 v8bf;
typedef __attribute__((ext_vector_type(8)))  float  v8f;
typedef __attribute__((ext_vector_type(4)))  int    v4i;

#define AS1 __attribute__((address_space(1)))
#define AS3 __attribute__((address_space(3)))

__device__ __forceinline__ v16bf cat8(v8bf a, v8bf b) {
  return __builtin_shufflevector(a, b, 0,1,2,3,4,5,6,7,8,9,10,11,12,13,14,15);
}

__device__ __forceinline__ v8f wmma_bf16(v16bf a, v16bf b, v8f c) {
  return __builtin_amdgcn_wmma_f32_16x16x32_bf16(false, a, false, b, (short)0, c,
                                                 false, false);
}

// Async global->LDS copy of 16 bytes (CDNA5 async path, tracked by ASYNCcnt).
// Toolchain builtin signature (from diagnostic): (AS1 v4i*, AS3 v4i*, Ii, Ii).
__device__ __forceinline__ void async_cp_b128(const bf16_t* g, bf16_t* l) {
#if __has_builtin(__builtin_amdgcn_global_load_async_to_lds_b128)
  __builtin_amdgcn_global_load_async_to_lds_b128(
      (AS1 v4i*)g, (AS3 v4i*)l, 0, 0);
#else
  unsigned lds = (unsigned)(size_t)((AS3 bf16_t*)l);
  unsigned long long ga = (unsigned long long)g;
  asm volatile("global_load_async_to_lds_b128 %0, %1, off"
               :: "v"(lds), "v"(ga) : "memory");
#endif
}

__device__ __forceinline__ void wait_async0() {
#if __has_builtin(__builtin_amdgcn_s_wait_asynccnt)
  __builtin_amdgcn_s_wait_asynccnt(0);
#else
  asm volatile("s_wait_asynccnt 0" ::: "memory");
#endif
}

// ---------------------------------------------------------------------------
// Problem constants
// ---------------------------------------------------------------------------
#define BB    32
#define HH    56
#define CC    384
#define NHD   12
#define WS_   7
#define SS_   3
#define NTOK  49            // WS*WS
#define NWIN  64            // (56/7)^2
#define HD_   32
#define MROWS 100352        // B * NWIN * NTOK
#define QKVC  1152
#define MLPH  1536

// ---------------------------------------------------------------------------
// Weight convert: fp32 W[K][N] -> bf16 W^T[N][K]  (so B-fragments stream
// contiguously per lane)
// ---------------------------------------------------------------------------
__global__ __launch_bounds__(256) void wconvT_k(const float* __restrict__ w,
                                                bf16_t* __restrict__ wt,
                                                int K, int N, int total) {
  int id = blockIdx.x * 256 + threadIdx.x;
  if (id >= total) return;
  int n = id / K;
  int k = id - n * K;
  wt[id] = (bf16_t)w[(size_t)k * N + n];
}

// ---------------------------------------------------------------------------
// LayerNorm (+ optional cyclic-shift + window partition) -> bf16 rows
// One block (128 threads) per output token row; C = 384 = 3 * 128.
// ---------------------------------------------------------------------------
template <bool PART>
__global__ __launch_bounds__(128) void ln_k(const float* __restrict__ x,
                                            const float* __restrict__ g,
                                            const float* __restrict__ b,
                                            bf16_t* __restrict__ out) {
  int row_out = blockIdx.x;
  size_t in_row;
  if (PART) {
    int wb  = row_out / NTOK;
    int tok = row_out - wb * NTOK;
    int bb  = wb >> 6;
    int w   = wb & 63;
    int wr = w >> 3, wc = w & 7;
    int i = tok / WS_, j = tok - i * WS_;
    int hs = wr * WS_ + i, ws = wc * WS_ + j;
    int hi = (hs + SS_) % HH, wi = (ws + SS_) % HH;   // roll(-3)
    in_row = (size_t)bb * (HH * HH) + (size_t)hi * HH + wi;
  } else {
    in_row = (size_t)row_out;
  }
  const float* px = x + in_row * CC;
  int t = threadIdx.x;
  float v0 = px[t], v1 = px[t + 128], v2 = px[t + 256];

  __shared__ float red[128];
  red[t] = v0 + v1 + v2;
  __syncthreads();
  for (int o = 64; o > 0; o >>= 1) {
    if (t < o) red[t] += red[t + o];
    __syncthreads();
  }
  float mean = red[0] * (1.0f / CC);
  __syncthreads();
  float d0 = v0 - mean, d1 = v1 - mean, d2 = v2 - mean;
  red[t] = d0 * d0 + d1 * d1 + d2 * d2;
  __syncthreads();
  for (int o = 64; o > 0; o >>= 1) {
    if (t < o) red[t] += red[t + o];
    __syncthreads();
  }
  float rstd = rsqrtf(red[0] * (1.0f / CC) + 1e-5f);

  bf16_t* po = out + (size_t)row_out * CC;
  po[t]       = (bf16_t)(d0 * rstd * g[t]       + b[t]);
  po[t + 128] = (bf16_t)(d1 * rstd * g[t + 128] + b[t + 128]);
  po[t + 256] = (bf16_t)(d2 * rstd * g[t + 256] + b[t + 256]);
}

// ---------------------------------------------------------------------------
// Window-reverse + un-shift destination row for the proj epilogue
// ---------------------------------------------------------------------------
__device__ __forceinline__ size_t dest_row(int m) {
  int wb  = m / NTOK;
  int tok = m - wb * NTOK;
  int bb  = wb >> 6;
  int w   = wb & 63;
  int wr = w >> 3, wc = w & 7;
  int i = tok / WS_, j = tok - i * WS_;
  int hs = wr * WS_ + i, ws = wc * WS_ + j;
  int ho = (hs + SS_) % HH, wo = (ws + SS_) % HH;     // roll(+3) inverse map
  return (size_t)bb * (HH * HH) + (size_t)ho * HH + wo;
}

// ---------------------------------------------------------------------------
// bf16 WMMA GEMM: C[M x Nout] = A[M x K] * W^T[Nout x K] (+bias, +epilogue)
// Block = 256 threads (8 waves). Block tile = 64 rows x 128 cols.
// Wave w owns the 16-col strip n0 = blockIdx.y*128 + w*16 and computes all
// four 16-row tiles (32 accumulator VGPRs). The shared 64x32 A-panel is
// staged once per k-step with async global->LDS (one b128 per thread) and
// shared by all 8 waves, so each B-fragment load feeds 4 WMMAs.
// EPI: 0 = bias -> bf16 store          1 = bias + exact GELU -> bf16
//      2 = bias + window-reverse scatter + residual -> f32 (writes x_res)
//      3 = bias + accumulate into f32 out (final residual add)
// ---------------------------------------------------------------------------
template <int EPI>
__global__ __launch_bounds__(256) void gemm_bf16_k(
    const bf16_t* __restrict__ A, const bf16_t* __restrict__ WT,
    const float* __restrict__ bias,
    bf16_t* __restrict__ obf, float* __restrict__ of32,
    const float* __restrict__ resid, int K, int Nout) {
  const int m0   = blockIdx.x * 64;
  const int tid  = threadIdx.x;
  const int wave = tid >> 5;
  const int lane = tid & 31;
  const int n0   = blockIdx.y * 128 + wave * 16;
  const int rlo  = lane & 15;
  const int k0   = (lane >> 4) * 8;

  __shared__ __align__(16) bf16_t At[64 * 32];   // 64-row x 32-col A panel

  const int srow = tid >> 2;        // 0..63  staging row
  const int sseg = tid & 3;         // 0..3   16B segment within row

  v8f acc[4] = {};
  for (int kk = 0; kk < K; kk += 32) {
    // stage A panel: 256 threads x 16B = 64x32 bf16
    async_cp_b128(A + (size_t)(m0 + srow) * K + kk + sseg * 8,
                  &At[srow * 32 + sseg * 8]);
    wait_async0();
    __syncthreads();

    const bf16_t* pb = WT + (size_t)(n0 + rlo) * K + kk + k0;
    v16bf bfr = cat8(*(const v8bf*)pb, *(const v8bf*)(pb + 16));

#pragma unroll
    for (int mt = 0; mt < 4; ++mt) {
      const bf16_t* pa = &At[(mt * 16 + rlo) * 32 + k0];
      v16bf af = cat8(*(const v8bf*)pa, *(const v8bf*)(pa + 16));
      acc[mt] = wmma_bf16(af, bfr, acc[mt]);
    }
    __syncthreads();
  }

  const int col = n0 + rlo;
  const float bc = bias[col];
#pragma unroll
  for (int mt = 0; mt < 4; ++mt) {
#pragma unroll
    for (int i = 0; i < 8; ++i) {
      int m = m0 + mt * 16 + i + (lane >> 4) * 8;  // C row = i + 8*(lane>=16)
      float v = acc[mt][i] + bc;
      if (EPI == 0) {
        obf[(size_t)m * Nout + col] = (bf16_t)v;
      } else if (EPI == 1) {
        float gel = 0.5f * v * (1.0f + erff(v * 0.70710678118654752f));
        obf[(size_t)m * Nout + col] = (bf16_t)gel;
      } else if (EPI == 2) {
        size_t d = dest_row(m);
        of32[d * CC + col] = resid[d * CC + col] + v;
      } else {
        of32[(size_t)m * Nout + col] += v;
      }
    }
  }
}

// ---------------------------------------------------------------------------
// Windowed attention: one block per (window-batch, head); 4 waves.
// Wave t owns the 16-row strip [16t, 16t+16) of the (padded) 64x64 score
// matrix. Head dim 32 => single K-step for Q*K^T. Softmax fused with
// analytic relative-position bias + shift mask. P restaged via LDS into
// A-fragment layout; V^T staged once per block in LDS for B-fragments.
// ---------------------------------------------------------------------------
__device__ __forceinline__ int reg3(int v) {
  return v < (HH - WS_) ? 0 : (v < (HH - SS_) ? 1 : 2);
}

__global__ __launch_bounds__(128) void attn_k(const bf16_t* __restrict__ qkv,
                                              const float* __restrict__ rpb,
                                              bf16_t* __restrict__ aout) {
  const int wb   = blockIdx.x;   // 0..2047
  const int h    = blockIdx.y;   // 0..11
  const int tid  = threadIdx.x;
  const int wave = tid >> 5;
  const int lane = tid & 31;
  const int rlo  = lane & 15;
  const int kh   = lane >> 4;
  const int k0   = kh * 8;

  __shared__ __align__(16) bf16_t vt[32 * 64];        // V^T: [hd][token]
  __shared__ __align__(16) bf16_t pst[4 * 16 * 64];   // per-wave P strips

  // Stage V^T (zero-pad tokens 49..63)
  for (int idx = tid; idx < 64 * 32; idx += 128) {
    int tok = idx >> 5, hd = idx & 31;
    float v = 0.0f;
    if (tok < NTOK)
      v = (float)qkv[(size_t)(wb * NTOK + tok) * QKVC + 2 * CC + h * HD_ + hd];
    vt[hd * 64 + tok] = (bf16_t)v;
  }
  __syncthreads();

  const int t = wave;

  // Q fragment (A layout): row = rlo, K chunks {k0..k0+7, k0+16..k0+23}
  v16bf aq;
  {
    v8bf c0 = {}, c1 = {};
    int tok = t * 16 + rlo;
    if (tok < NTOK) {
      const bf16_t* p = qkv + (size_t)(wb * NTOK + tok) * QKVC + h * HD_;
      c0 = *(const v8bf*)(p + k0);
      c1 = *(const v8bf*)(p + k0 + 16);
    }
    aq = cat8(c0, c1);
  }

  // S strip = Q (16x32) * K^T (32x64): 4 WMMAs, B-frag(u) = K-rows 16u..16u+15
  v8f S[4];
#pragma unroll
  for (int u = 0; u < 4; ++u) {
    v8bf c0 = {}, c1 = {};
    int krow = u * 16 + rlo;
    if (krow < NTOK) {
      const bf16_t* p = qkv + (size_t)(wb * NTOK + krow) * QKVC + CC + h * HD_;
      c0 = *(const v8bf*)(p + k0);
      c1 = *(const v8bf*)(p + k0 + 16);
    }
    v8f z = {};
    S[u] = wmma_bf16(aq, cat8(c0, c1), z);
  }

  // Fused scale + rel-pos bias + shift mask + softmax
  const int w  = wb & 63;
  const int wr = w >> 3, wc = w & 7;
  const float scale = 0.17677669529663687f;  // 32^-0.5
  float ex[4][8];

#pragma unroll
  for (int i = 0; i < 8; ++i) {
    int nIdx = t * 16 + i + kh * 8;
    int r1 = nIdx / WS_, c1 = nIdx - r1 * WS_;
    int regn = reg3(wr * WS_ + r1) * 3 + reg3(wc * WS_ + c1);
#pragma unroll
    for (int u = 0; u < 4; ++u) {
      int mIdx = u * 16 + rlo;
      float s = S[u][i] * scale;
      if (nIdx < NTOK && mIdx < NTOK) {
        int r2 = mIdx / WS_, c2 = mIdx - r2 * WS_;
        int rpi = (r1 - r2 + WS_ - 1) * (2 * WS_ - 1) + (c1 - c2 + WS_ - 1);
        s += rpb[rpi * NHD + h];
        int regm = reg3(wr * WS_ + r2) * 3 + reg3(wc * WS_ + c2);
        if (regn != regm) s += -100.0f;
      } else {
        s = (nIdx < NTOK) ? -1e30f : 0.0f;  // pad cols -> -inf; pad rows benign
      }
      ex[u][i] = s;
    }
  }

#pragma unroll
  for (int i = 0; i < 8; ++i) {
    float m = ex[0][i];
#pragma unroll
    for (int u = 1; u < 4; ++u) m = fmaxf(m, ex[u][i]);
    // row lives in one 16-lane half: xor-reduce over bits 0..3 (bit4 fixed)
    m = fmaxf(m, __shfl_xor(m, 1));
    m = fmaxf(m, __shfl_xor(m, 2));
    m = fmaxf(m, __shfl_xor(m, 4));
    m = fmaxf(m, __shfl_xor(m, 8));
    float sum = 0.0f;
#pragma unroll
    for (int u = 0; u < 4; ++u) {
      ex[u][i] = __expf(ex[u][i] - m);
      sum += ex[u][i];
    }
    sum += __shfl_xor(sum, 1);
    sum += __shfl_xor(sum, 2);
    sum += __shfl_xor(sum, 4);
    sum += __shfl_xor(sum, 8);
    float rinv = __frcp_rn(sum);
    int prow = i + kh * 8;
#pragma unroll
    for (int u = 0; u < 4; ++u)
      pst[t * 16 * 64 + prow * 64 + u * 16 + rlo] = (bf16_t)(ex[u][i] * rinv);
  }

  // O strip (16x32) = P (16x64) * V (64x32): 2 N-tiles x 2 K-steps
#pragma unroll
  for (int n2 = 0; n2 < 2; ++n2) {
    v8f acc = {};
#pragma unroll
    for (int s = 0; s < 2; ++s) {
      int kk = s * 32 + k0;
      const bf16_t* pa = &pst[t * 16 * 64 + rlo * 64 + kk];
      v16bf af = cat8(*(const v8bf*)pa, *(const v8bf*)(pa + 16));
      const bf16_t* pb = &vt[(n2 * 16 + rlo) * 64 + kk];
      v16bf bfr = cat8(*(const v8bf*)pb, *(const v8bf*)(pb + 16));
      acc = wmma_bf16(af, bfr, acc);
    }
#pragma unroll
    for (int i = 0; i < 8; ++i) {
      int tok = t * 16 + i + kh * 8;
      if (tok < NTOK)
        aout[(size_t)(wb * NTOK + tok) * CC + h * HD_ + n2 * 16 + rlo] =
            (bf16_t)acc[i];
    }
  }
}

// ---------------------------------------------------------------------------
// Host launcher
// ---------------------------------------------------------------------------
extern "C" void kernel_launch(void* const* d_in, const int* in_sizes, int n_in,
                              void* d_out, int out_size, void* d_ws,
                              size_t ws_size, hipStream_t stream) {
  const float* x     = (const float*)d_in[0];
  const float* n1w   = (const float*)d_in[1];
  const float* n1b   = (const float*)d_in[2];
  const float* qkvw  = (const float*)d_in[3];
  const float* qkvb  = (const float*)d_in[4];
  const float* rpb   = (const float*)d_in[5];
  const float* projw = (const float*)d_in[6];
  const float* projb = (const float*)d_in[7];
  const float* n2w   = (const float*)d_in[8];
  const float* n2b   = (const float*)d_in[9];
  const float* fc1w  = (const float*)d_in[10];
  const float* fc1b  = (const float*)d_in[11];
  const float* fc2w  = (const float*)d_in[12];
  const float* fc2b  = (const float*)d_in[13];
  float* out = (float*)d_out;

  char* ws = (char*)d_ws;
  size_t off = 0;
  auto take = [&](size_t bytes) {
    size_t r = off;
    off = (off + bytes + 255) & ~(size_t)255;
    return r;
  };
  bf16_t* qkvT = (bf16_t*)(ws + take((size_t)QKVC * CC * 2));
  bf16_t* projT = (bf16_t*)(ws + take((size_t)CC * CC * 2));
  bf16_t* fc1T = (bf16_t*)(ws + take((size_t)MLPH * CC * 2));
  bf16_t* fc2T = (bf16_t*)(ws + take((size_t)CC * MLPH * 2));
  bf16_t* xw   = (bf16_t*)(ws + take((size_t)MROWS * CC * 2));   // xn / ln2
  bf16_t* ao   = (bf16_t*)(ws + take((size_t)MROWS * CC * 2));   // attn out
  bf16_t* big  = (bf16_t*)(ws + take((size_t)MROWS * MLPH * 2)); // qkv, then h1
  bf16_t* qkvact = big;   // MROWS x 1152 (dead before fc1 writes h1)
  bf16_t* h1     = big;   // MROWS x 1536

  // 1) weights -> bf16 transposed
  wconvT_k<<<dim3((QKVC * CC + 255) / 256), 256, 0, stream>>>(qkvw, qkvT, CC, QKVC, QKVC * CC);
  wconvT_k<<<dim3((CC * CC + 255) / 256), 256, 0, stream>>>(projw, projT, CC, CC, CC * CC);
  wconvT_k<<<dim3((MLPH * CC + 255) / 256), 256, 0, stream>>>(fc1w, fc1T, CC, MLPH, MLPH * CC);
  wconvT_k<<<dim3((CC * MLPH + 255) / 256), 256, 0, stream>>>(fc2w, fc2T, MLPH, CC, CC * MLPH);

  // 2) LN1 + shift + window partition -> bf16
  ln_k<true><<<dim3(MROWS), 128, 0, stream>>>(x, n1w, n1b, xw);

  // 3) QKV GEMM (100352 x 384 x 1152), 64x128 block tiles
  gemm_bf16_k<0><<<dim3(MROWS / 64, QKVC / 128), 256, 0, stream>>>(
      xw, qkvT, qkvb, qkvact, nullptr, nullptr, CC, QKVC);

  // 4) windowed attention (2048 windows x 12 heads)
  attn_k<<<dim3(BB * NWIN, NHD), 128, 0, stream>>>(qkvact, rpb, ao);

  // 5) proj GEMM + window-reverse + unshift + residual -> d_out (= x_res)
  gemm_bf16_k<2><<<dim3(MROWS / 64, CC / 128), 256, 0, stream>>>(
      ao, projT, projb, nullptr, out, x, CC, CC);

  // 6) LN2 on x_res -> bf16
  ln_k<false><<<dim3(MROWS), 128, 0, stream>>>(out, n2w, n2b, xw);

  // 7) fc1 GEMM + exact GELU (100352 x 384 x 1536)
  gemm_bf16_k<1><<<dim3(MROWS / 64, MLPH / 128), 256, 0, stream>>>(
      xw, fc1T, fc1b, h1, nullptr, nullptr, CC, MLPH);

  // 8) fc2 GEMM + residual accumulate into d_out (100352 x 1536 x 384)
  gemm_bf16_k<3><<<dim3(MROWS / 64, CC / 128), 256, 0, stream>>>(
      h1, fc2T, fc2b, nullptr, out, nullptr, MLPH, CC);
}
